// LSTMModel_5205500362875
// MI455X (gfx1250) — matrix operand order repaired and checked by reading, not compiled, not run on hardware
//
#include <hip/hip_runtime.h>
#include <hip/hip_bf16.h>
#include <math.h>

// ---------------------------------------------------------------------------
// Fused 2-layer LSTM (B=4096, T=512, H=50, IN=1) + final FC, single kernel.
// Each wave32 owns 16 batch rows for the entire time loop.
// Weights pre-packed in LDS as bf16 WMMA B-fragments; gates accumulated in
// f32 via v_wmma_f32_16x16x32_bf16; cell state f32 in VGPRs.
// ---------------------------------------------------------------------------

typedef __attribute__((ext_vector_type(16))) __bf16 v16bf;
typedef __attribute__((ext_vector_type(8)))  __bf16 v8bf;
typedef __attribute__((ext_vector_type(8)))  float  v8f;

#define B_    4096
#define T_    512
#define H_    50
#define G4_   200      // 4*H gate width
#define NT_   13       // N tiles of 16  (208 padded)
#define NP_   208
#define GSTR  212      // gate-staging row stride in floats (bank-conflict pad)
#define HSTR  72       // h row stride in bf16 (bank-conflict pad, 16B aligned)
#define WAVES 4
#define ROWS  16       // batch rows per wave

__device__ __forceinline__ v8f wmma_bf16(v16bf a, v16bf b, v8f c) {
  // (neg_a, A, neg_b, B, c_mod, C, reuse_a, reuse_b)
  return __builtin_amdgcn_wmma_f32_16x16x32_bf16(false, a, false, b,
                                                 (short)0, c, false, false);
}

__device__ __forceinline__ float sigf(float x) {
  return 1.0f / (1.0f + __expf(-x));
}

// A-fragment (16x32 bf16, M=batch rows, K=hidden) from LDS h tile.
// ISA layout: lane L (L<16): M=L, K = {0..7, 16..23}; L>=16: M=L-16, K = {8..15, 24..31}.
__device__ __forceinline__ v16bf loadA(const __bf16* hrow, int lane, int kc) {
  const int m  = lane & 15;
  const int kb = (lane >> 4) * 8;
  const __bf16* p = hrow + m * HSTR + kc * 32 + kb;
  v8bf lo = *(const v8bf*)p;          // K = kc*32 + kb .. +7
  v8bf hi = *(const v8bf*)(p + 16);   // K = kc*32 + 16 + kb .. +7
  return __builtin_shufflevector(lo, hi, 0,1,2,3,4,5,6,7,8,9,10,11,12,13,14,15);
}

__global__ __launch_bounds__(WAVES * 32)
void lstm2_fused_kernel(const float* __restrict__ x,
                        const float* __restrict__ Wih0, const float* __restrict__ Whh0,
                        const float* __restrict__ bih0, const float* __restrict__ bhh0,
                        const float* __restrict__ Wih1, const float* __restrict__ Whh1,
                        const float* __restrict__ bih1, const float* __restrict__ bhh1,
                        const float* __restrict__ Wfc,  const float* __restrict__ bfc,
                        float* __restrict__ out)
{
  // Packed B-fragments: mat 0 = Whh0, 1 = Wih1, 2 = Whh1.
  // Fragment (mat, nt, kc): 32 lanes x 16 bf16 (contiguous per lane -> ds_load_b128 x2).
  __shared__ __attribute__((aligned(64))) __bf16 wPack[3 * NT_ * 2 * 512]; // 79.9 KB
  __shared__ __attribute__((aligned(64))) float  gbuf[WAVES][ROWS * GSTR]; // 54.3 KB
  __shared__ __attribute__((aligned(64))) __bf16 hA[WAVES][2][ROWS * HSTR]; // 18.4 KB
  __shared__ float xbuf[WAVES][ROWS];
  __shared__ float b0s[NP_], wcol0[NP_], b1s[NP_], wfcS[64];

  const int tid = threadIdx.x;

  // ---- one-time pack: weights -> LDS bf16 B-fragments ------------------
  // B is 32x16 (KxN): lane L (L<16): N=L, K = {0..7, 16..23};
  // L>=16: N=L-16, K = {8..15, 24..31}.  Element value = W[n][k] (h @ W.T).
  for (int idx = tid; idx < 3 * NT_ * 2 * 512; idx += WAVES * 32) {
    const int e    = idx & 15;
    const int lane = (idx >> 4) & 31;
    const int kc   = (idx >> 9) & 1;
    const int nt   = (idx >> 10) % NT_;
    const int mat  = (idx >> 10) / NT_;
    const int n = nt * 16 + (lane & 15);
    const int k = kc * 32 + ((lane >> 4) << 3) + (e & 7) + ((e >> 3) << 4);
    const float* W = (mat == 0) ? Whh0 : ((mat == 1) ? Wih1 : Whh1);
    float v = 0.0f;
    if (n < G4_ && k < H_) v = W[n * H_ + k];
    wPack[idx] = (__bf16)v;
  }
  for (int i = tid; i < NP_; i += WAVES * 32) {
    b0s[i]   = (i < G4_) ? (bih0[i] + bhh0[i]) : 0.0f;
    wcol0[i] = (i < G4_) ? Wih0[i]             : 0.0f;   // Wih0 is [200,1]
    b1s[i]   = (i < G4_) ? (bih1[i] + bhh1[i]) : 0.0f;
  }
  for (int i = tid; i < 64; i += WAVES * 32) wfcS[i] = (i < H_) ? Wfc[i] : 0.0f;
  for (int i = tid; i < WAVES * 2 * ROWS * HSTR; i += WAVES * 32)
    ((__bf16*)hA)[i] = (__bf16)0.0f;                     // h0 = 0, K-padding = 0
  __syncthreads();

  const int wave    = tid >> 5;
  const int lane    = tid & 31;
  const int lane16  = lane & 15;
  const int mb      = (lane >> 4) * 8;   // C-fragment M base for this lane half
  const int rowBase = blockIdx.x * (WAVES * ROWS) + wave * ROWS;

  float c0s[25], c1s[25], hst[25];
#pragma unroll
  for (int s = 0; s < 25; ++s) { c0s[s] = 0.0f; c1s[s] = 0.0f; hst[s] = 0.0f; }

  float*  gw  = gbuf[wave];
  __bf16* h0p = hA[wave][0];
  __bf16* h1p = hA[wave][1];

#pragma clang loop unroll(disable)
  for (int t = 0; t < T_; ++t) {
    if (lane < ROWS) {
      xbuf[wave][lane] = x[(rowBase + lane) * T_ + t];
      __builtin_prefetch(&x[(rowBase + lane) * T_ + t + 16], 0, 1);
    }

    // ------------------------- layer 0 ------------------------------
    v16bf a00 = loadA(h0p, lane, 0);
    v16bf a01 = loadA(h0p, lane, 1);
#pragma unroll
    for (int nt = 0; nt < NT_; ++nt) {
      v8f acc = {};
      acc = wmma_bf16(a00, *(const v16bf*)(wPack + ((0 * NT_ + nt) * 2 + 0) * 512 + lane * 16), acc);
      acc = wmma_bf16(a01, *(const v16bf*)(wPack + ((0 * NT_ + nt) * 2 + 1) * 512 + lane * 16), acc);
      const int   n  = nt * 16 + lane16;
      const float wc = wcol0[n];
      const float bb = b0s[n];
#pragma unroll
      for (int v = 0; v < 8; ++v) {
        const int m = mb + v;
        gw[m * GSTR + n] = acc[v] + xbuf[wave][m] * wc + bb;  // IN==1 folded here
      }
    }
    // gate nonlinearities + cell update (same-wave LDS ops are in-order)
#pragma unroll
    for (int s = 0; s < 25; ++s) {
      const int p = s * 32 + lane;                // 0..799 = m*50 + hh
      const int m = p / 50, hh = p - m * 50;
      const float* gr = gw + m * GSTR;
      const float gi = sigf (gr[hh]);
      const float gf = sigf (gr[50  + hh]);
      const float gg = tanhf(gr[100 + hh]);
      const float go = sigf (gr[150 + hh]);
      const float c  = gf * c0s[s] + gi * gg;
      c0s[s] = c;
      h0p[m * HSTR + hh] = (__bf16)(go * tanhf(c));
    }

    // ------------------------- layer 1 ------------------------------
    v16bf ar0 = loadA(h1p, lane, 0);   // recurrent h1_{t-1}
    v16bf ar1 = loadA(h1p, lane, 1);
    v16bf ai0 = loadA(h0p, lane, 0);   // fresh h0_t just written above
    v16bf ai1 = loadA(h0p, lane, 1);
#pragma unroll
    for (int nt = 0; nt < NT_; ++nt) {
      v8f acc = {};
      acc = wmma_bf16(ai0, *(const v16bf*)(wPack + ((1 * NT_ + nt) * 2 + 0) * 512 + lane * 16), acc);
      acc = wmma_bf16(ai1, *(const v16bf*)(wPack + ((1 * NT_ + nt) * 2 + 1) * 512 + lane * 16), acc);
      acc = wmma_bf16(ar0, *(const v16bf*)(wPack + ((2 * NT_ + nt) * 2 + 0) * 512 + lane * 16), acc);
      acc = wmma_bf16(ar1, *(const v16bf*)(wPack + ((2 * NT_ + nt) * 2 + 1) * 512 + lane * 16), acc);
      const int   n  = nt * 16 + lane16;
      const float bb = b1s[n];
#pragma unroll
      for (int v = 0; v < 8; ++v)
        gw[(mb + v) * GSTR + n] = acc[v] + bb;
    }
#pragma unroll
    for (int s = 0; s < 25; ++s) {
      const int p = s * 32 + lane;
      const int m = p / 50, hh = p - m * 50;
      const float* gr = gw + m * GSTR;
      const float gi = sigf (gr[hh]);
      const float gf = sigf (gr[50  + hh]);
      const float gg = tanhf(gr[100 + hh]);
      const float go = sigf (gr[150 + hh]);
      const float c  = gf * c1s[s] + gi * gg;
      c1s[s] = c;
      const float hn = go * tanhf(c);
      hst[s] = hn;                       // keep f32 copy for the FC head
      h1p[m * HSTR + hh] = (__bf16)hn;
    }
  }

  // ------------------- FC head on final h2 (f32) ----------------------
#pragma unroll
  for (int s = 0; s < 25; ++s)
    gw[s * 32 + lane] = hst[s];          // index == m*50 + hh (exact)
  __builtin_amdgcn_wave_barrier();       // keep store->load order across lanes
  if (lane < ROWS) {
    float acc = bfc[0];
#pragma unroll 10
    for (int hh = 0; hh < H_; ++hh)
      acc += gw[lane * 50 + hh] * wfcS[hh];
    out[rowBase + lane] = acc;
  }
}

extern "C" void kernel_launch(void* const* d_in, const int* in_sizes, int n_in,
                              void* d_out, int out_size, void* d_ws, size_t ws_size,
                              hipStream_t stream) {
  const float* x    = (const float*)d_in[0];
  const float* Wih0 = (const float*)d_in[1];
  const float* Whh0 = (const float*)d_in[2];
  const float* bih0 = (const float*)d_in[3];
  const float* bhh0 = (const float*)d_in[4];
  const float* Wih1 = (const float*)d_in[5];
  const float* Whh1 = (const float*)d_in[6];
  const float* bih1 = (const float*)d_in[7];
  const float* bhh1 = (const float*)d_in[8];
  const float* Wfc  = (const float*)d_in[9];
  const float* bfc  = (const float*)d_in[10];
  float* out = (float*)d_out;

  dim3 grid(B_ / (WAVES * ROWS));   // 64 workgroups
  dim3 block(WAVES * 32);           // 128 threads = 4 waves
  lstm2_fused_kernel<<<grid, block, 0, stream>>>(
      x, Wih0, Whh0, bih0, bhh0, Wih1, Whh1, bih1, bhh1, Wfc, bfc, out);
}
